// ActorBatchNet_49546742726688
// MI455X (gfx1250) — matrix operand order repaired and checked by reading, not compiled, not run on hardware
//
#include <hip/hip_runtime.h>
#include <hip/hip_fp16.h>

typedef __attribute__((ext_vector_type(16))) _Float16 v16h;
typedef __attribute__((ext_vector_type(8)))  float    v8f;

static __device__ __forceinline__ float sigm(float x) { return 1.0f / (1.0f + expf(-x)); }

// ---------------------------------------------------------------- utilities
__global__ void k_zero(float* __restrict__ p, long n) {
    long i = (long)blockIdx.x * blockDim.x + threadIdx.x;
    if (i < n) p[i] = 0.0f;
}

// f32 -> f16 weight convert; trans: dst[c*R + r] = src[r*C + c]
__global__ void k_cvt(const float* __restrict__ s, _Float16* __restrict__ d,
                      int R, int C, int trans) {
    int i = blockIdx.x * blockDim.x + threadIdx.x;
    if (i >= R * C) return;
    int r = i / C, c = i % C;
    if (trans) d[(long)c * R + r] = (_Float16)s[i];
    else       d[i] = (_Float16)s[i];
}

// ---------------------------------------------------------------- WMMA GEMM
// C[M,Ncols] = A[M,K] (f16) * W[Ncols,K]^T (f16)  + bias[Ncols], opt relu.
// One wave computes one 16-row M tile x FOUR 16-col N tiles (A-frag reused 4x).
// K stepped by 32 via v_wmma_f32_16x16x32_f16.
// Fragment layouts per CDNA5 ISA 7.12.2 (wave32):
//   A 16x32: lane m=l&15, kh=l>>4; half j: K = (j<8 ? j : 16+(j-8)) + kh*8
//   B 32x16: lane n=l&15, kh=l>>4; half j: K = (j<8 ? j : 16+(j-8)) + kh*8
//   C 16x16: lane n=l&15; vgpr i:  M = i + (l>>4)*8
// Requires M%16==0, Ncols%64==0, K%32==0.
template <bool F16OUT>
__global__ void gemm_wmma4(const _Float16* __restrict__ A,
                           const _Float16* __restrict__ W,
                           const float* __restrict__ bias,
                           void* __restrict__ C,
                           int M, int Ncols, int K, int relu) {
    int wave = (int)((blockIdx.x * (long)blockDim.x + threadIdx.x) >> 5);
    int lane = threadIdx.x & 31;
    int groupsN = Ncols >> 6;                 // groups of 4 N-tiles
    int tm = wave / groupsN;
    int gn = wave - tm * groupsN;
    if (tm * 16 >= M) return;                 // wave-uniform: EXEC stays all-1s
    int mr = lane & 15;
    int kh = lane >> 4;
    v8f acc[4] = {{}, {}, {}, {}};
    const _Float16* arow  = A + (size_t)(tm * 16 + mr) * K;
    const _Float16* wbase = W + (size_t)(gn * 64 + mr) * K;
    for (int k0 = 0; k0 < K; k0 += 32) {
        v16h a;
#pragma unroll
        for (int j = 0; j < 8; ++j)  a[j]     = arow[k0 + kh * 8 + j];
#pragma unroll
        for (int j = 0; j < 8; ++j)  a[8 + j] = arow[k0 + 16 + kh * 8 + j];
#pragma unroll
        for (int t = 0; t < 4; ++t) {
            v16h b;
            const _Float16* wrow = wbase + (size_t)t * 16 * K;
#pragma unroll
            for (int j = 0; j < 16; ++j) b[j] = wrow[k0 + (j < 8 ? j : 16 + (j - 8)) + kh * 8];
            acc[t] = __builtin_amdgcn_wmma_f32_16x16x32_f16(false, a, false, b,
                                                            (short)0, acc[t], false, false);
        }
    }
#pragma unroll
    for (int t = 0; t < 4; ++t) {
        int n = gn * 64 + t * 16 + mr;
        float bv = bias ? bias[n] : 0.0f;
#pragma unroll
        for (int i = 0; i < 8; ++i) {
            int m = tm * 16 + kh * 8 + i;
            float v = acc[t][i] + bv;
            if (relu) v = fmaxf(v, 0.0f);
            if (F16OUT) ((_Float16*)C)[(size_t)m * Ncols + n] = (_Float16)v;
            else        ((float*)C)[(size_t)m * Ncols + n] = v;
        }
    }
}

// ---------------------------------------------------------------- front end
// node = relu(x @ lin0_w^T + b)   (K=3, scalar)
__global__ void k_lin0(const float* __restrict__ x, const float* __restrict__ w,
                       const float* __restrict__ b, float* __restrict__ node,
                       _Float16* __restrict__ node_h, int N) {
    int i = blockIdx.x * blockDim.x + threadIdx.x;
    if (i >= N * 64) return;
    int n = i >> 6, o = i & 63;
    float acc = b[o] + x[n * 3 + 0] * w[o * 3 + 0]
                     + x[n * 3 + 1] * w[o * 3 + 1]
                     + x[n * 3 + 2] * w[o * 3 + 2];
    acc = fmaxf(acc, 0.0f);
    node[i] = acc;
    node_h[i] = (_Float16)acc;
}

// t = relu(edge_attr @ e1_w^T + b)   (K=6, scalar) -> f16
__global__ void k_e1(const float* __restrict__ ea, const float* __restrict__ w,
                     const float* __restrict__ b, _Float16* __restrict__ t_h, int E) {
    int i = blockIdx.x * blockDim.x + threadIdx.x;
    if (i >= E * 64) return;
    int e = i >> 6, o = i & 63;
    float acc = b[o];
#pragma unroll
    for (int k = 0; k < 6; ++k) acc += ea[e * 6 + k] * w[o * 6 + k];
    t_h[i] = (_Float16)fmaxf(acc, 0.0f);
}

__global__ void k_count(const int* __restrict__ dst, float* __restrict__ deg, int E) {
    int e = blockIdx.x * blockDim.x + threadIdx.x;
    if (e < E) atomicAdd(&deg[dst[e]], 1.0f);
}

// ---------------------------------------------------------------- message passing
// wave per edge: msg = node[src] (1x64) @ ew_e (64x64); scatter-add into agg[dst]
__global__ void k_msg(const float* __restrict__ node, const _Float16* __restrict__ ew,
                      const int* __restrict__ src, const int* __restrict__ dst,
                      float* __restrict__ agg, int E) {
    int wave = (int)((blockIdx.x * (long)blockDim.x + threadIdx.x) >> 5);
    int lane = threadIdx.x & 31;
    if (wave >= E) return;
    int s = src[wave], t = dst[wave];
    const _Float16* m = ew + (size_t)wave * 4096;
    const float* xr = node + (size_t)s * 64;
    int o = lane * 2;
    float a0 = 0.0f, a1 = 0.0f;
    for (int i = 0; i < 64; ++i) {
        float xv = xr[i];                 // broadcast across wave
        a0 += xv * (float)m[i * 64 + o];
        a1 += xv * (float)m[i * 64 + o + 1];
    }
    atomicAdd(&agg[(size_t)t * 64 + o], a0);
    atomicAdd(&agg[(size_t)t * 64 + o + 1], a1);
}

// m = relu(agg/deg + node@root_w + conv_b) -> f16
__global__ void k_m(const float* __restrict__ agg, const float* __restrict__ deg,
                    const float* __restrict__ rootout, const float* __restrict__ cb,
                    _Float16* __restrict__ m_h, int N) {
    int i = blockIdx.x * blockDim.x + threadIdx.x;
    if (i >= N * 64) return;
    int n = i >> 6, o = i & 63;
    float dg = fmaxf(deg[n], 1.0f);
    float v = agg[i] / dg + rootout[i] + cb[o];
    m_h[i] = (_Float16)fmaxf(v, 0.0f);
}

// GRU cell update (gi/gh hold x@Wih^T+bih and h@Whh^T+bhh)
__global__ void k_gru(const float* __restrict__ gi, const float* __restrict__ gh,
                      float* __restrict__ node, _Float16* __restrict__ node_h, int N) {
    int i = blockIdx.x * blockDim.x + threadIdx.x;
    if (i >= N * 64) return;
    int n = i >> 6, j = i & 63;
    const float* Gi = gi + (size_t)n * 192;
    const float* Gh = gh + (size_t)n * 192;
    float r  = sigm(Gi[j] + Gh[j]);
    float z  = sigm(Gi[64 + j] + Gh[64 + j]);
    float nn = tanhf(Gi[128 + j] + r * Gh[128 + j]);
    float hn = (1.0f - z) * nn + z * node[i];
    node[i] = hn;
    node_h[i] = (_Float16)hn;
}

// ---------------------------------------------------------------- Set2Set / LSTM
// one block (64 threads) per graph; gate order i,f,g,o
__global__ void k_lstm(const float* __restrict__ xin, int Kin,
                       const float* __restrict__ wih, const float* __restrict__ whh,
                       const float* __restrict__ bih, const float* __restrict__ bhh,
                       float* __restrict__ h, float* __restrict__ c, int G) {
    int g = blockIdx.x, j = threadIdx.x;
    __shared__ float hsh[64];
    hsh[j] = h[g * 64 + j];
    __syncthreads();
    float pre[4];
    for (int gate = 0; gate < 4; ++gate) {
        int row = gate * 64 + j;
        float acc = bih[row] + bhh[row];
        const float* wi = wih + (long)row * Kin;
        const float* xr = xin + (long)g * Kin;
        for (int k = 0; k < Kin; ++k) acc += xr[k] * wi[k];
        const float* wh = whh + (long)row * 64;
        for (int k = 0; k < 64; ++k) acc += hsh[k] * wh[k];
        pre[gate] = acc;
    }
    float cv = c[g * 64 + j];
    float cn = sigm(pre[1]) * cv + sigm(pre[0]) * tanhf(pre[2]);
    float hn = sigm(pre[3]) * tanhf(cn);
    c[g * 64 + j] = cn;
    h[g * 64 + j] = hn;
}

// per-graph softmax attention + weighted sum; writes q_star = [hs, r]
// block = npg (=64) threads per graph
__global__ void k_attend(const float* __restrict__ node, const float* __restrict__ hs,
                         float* __restrict__ qstar, int npg) {
    int g = blockIdx.x, t = threadIdx.x;
    __shared__ float red[64];
    __shared__ float av[64];
    const float* hg = hs + (size_t)g * 64;
    const float* nr = node + ((size_t)g * npg + t) * 64;
    float e = 0.0f;
    for (int k = 0; k < 64; ++k) e += nr[k] * hg[k];
    red[t] = e; __syncthreads();
    for (int s = 32; s > 0; s >>= 1) { if (t < s) red[t] = fmaxf(red[t], red[t + s]); __syncthreads(); }
    float emax = red[0]; __syncthreads();
    float a = expf(e - emax);
    red[t] = a; __syncthreads();
    for (int s = 32; s > 0; s >>= 1) { if (t < s) red[t] += red[t + s]; __syncthreads(); }
    float asum = red[0]; __syncthreads();
    a /= asum;
    av[t] = a; __syncthreads();
    float r = 0.0f;
    for (int n = 0; n < npg; ++n) r += av[n] * node[((size_t)g * npg + n) * 64 + t];
    qstar[g * 128 + t] = hg[t];
    qstar[g * 128 + 64 + t] = r;
}

// ---------------------------------------------------------------- head
// faithful [5,T,d] -> transpose(2,1,0) -> reshape(T, 5d) gather, f16 out
__global__ void k_feat(const float* __restrict__ node, const float* __restrict__ hx,
                       const int* __restrict__ nonring, const int* __restrict__ nrbidx,
                       _Float16* __restrict__ feat_h, int T) {
    long i = (long)blockIdx.x * blockDim.x + threadIdx.x;
    if (i >= (long)T * 320) return;
    long flat = i;                       // row-major over [T,320] == flat over [d,T,5]
    int dd = (int)(flat / ((long)T * 5));
    long rem = flat % ((long)T * 5);
    int tt = (int)(rem / 5);
    int s = (int)(rem % 5);
    float v;
    if (s == 0) v = hx[(size_t)nrbidx[tt] * 64 + dd];
    else        v = node[(size_t)nonring[(long)(s - 1) * T + tt] * 64 + dd];
    feat_h[i] = (_Float16)v;
}

__global__ void k_lin2(const float* __restrict__ o1, const float* __restrict__ w,
                       const float* __restrict__ b, float* __restrict__ out, int T) {
    int i = blockIdx.x * blockDim.x + threadIdx.x;
    if (i >= T * 6) return;
    int t = i / 6, a = i - t * 6;
    float acc = b[a];
    for (int k = 0; k < 64; ++k) acc += o1[(size_t)t * 64 + k] * w[a * 64 + k];
    out[i] = acc;
}

// ---------------------------------------------------------------- launcher
extern "C" void kernel_launch(void* const* d_in, const int* in_sizes, int n_in,
                              void* d_out, int out_size, void* d_ws, size_t ws_size,
                              hipStream_t stream) {
    const float* x         = (const float*)d_in[0];
    const float* edge_attr = (const float*)d_in[1];
    const int*   edge_idx  = (const int*)d_in[2];
    const int*   nonring   = (const int*)d_in[4];
    const int*   nrbidx    = (const int*)d_in[5];
    const float* lin0_w = (const float*)d_in[8];
    const float* lin0_b = (const float*)d_in[9];
    const float* e1_w   = (const float*)d_in[10];
    const float* e1_b   = (const float*)d_in[11];
    const float* e2_w   = (const float*)d_in[12];
    const float* e2_b   = (const float*)d_in[13];
    const float* root_w = (const float*)d_in[14];
    const float* conv_b = (const float*)d_in[15];
    const float* gru_wih = (const float*)d_in[16];
    const float* gru_whh = (const float*)d_in[17];
    const float* gru_bih = (const float*)d_in[18];
    const float* gru_bhh = (const float*)d_in[19];
    const float* s2s_wih = (const float*)d_in[20];
    const float* s2s_whh = (const float*)d_in[21];
    const float* s2s_bih = (const float*)d_in[22];
    const float* s2s_bhh = (const float*)d_in[23];
    const float* mem_wih = (const float*)d_in[24];
    const float* mem_whh = (const float*)d_in[25];
    const float* mem_bih = (const float*)d_in[26];
    const float* mem_bhh = (const float*)d_in[27];
    const float* lin1_w = (const float*)d_in[28];
    const float* lin1_b = (const float*)d_in[29];
    const float* lin2_w = (const float*)d_in[30];
    const float* lin2_b = (const float*)d_in[31];

    const int N = in_sizes[0] / 3;
    const int E = in_sizes[1] / 6;
    const int T = in_sizes[5];                // G * TPG
    const int G = (out_size - 6 * T) / 128;   // out = 6T + 2*G*64
    const int npg = N / G;                    // nodes per graph (contiguous batch)

    const int* srcE = edge_idx;
    const int* dstE = edge_idx + E;

    char* w = (char*)d_ws;
    size_t off = 0;
    auto alloc = [&](size_t bytes) -> void* {
        void* p = w + off; off += (bytes + 255) & ~(size_t)255; return p;
    };
    _Float16* ew_h    = (_Float16*)alloc((size_t)E * 4096 * 2);  // 256 MB
    _Float16* t_h     = (_Float16*)alloc((size_t)E * 64 * 2);
    float*    node    = (float*)   alloc((size_t)N * 64 * 4);
    _Float16* node_h  = (_Float16*)alloc((size_t)N * 64 * 2);
    float*    agg     = (float*)   alloc((size_t)N * 64 * 4);
    float*    rootout = (float*)   alloc((size_t)N * 64 * 4);
    _Float16* m_h     = (_Float16*)alloc((size_t)N * 64 * 2);
    float*    gi      = (float*)   alloc((size_t)N * 192 * 4);
    float*    gh      = (float*)   alloc((size_t)N * 192 * 4);
    float*    deg     = (float*)   alloc((size_t)N * 4);
    _Float16* e2w_h   = (_Float16*)alloc((size_t)4096 * 64 * 2);
    _Float16* rootT_h = (_Float16*)alloc((size_t)64 * 64 * 2);
    _Float16* wih_h   = (_Float16*)alloc((size_t)192 * 64 * 2);
    _Float16* whh_h   = (_Float16*)alloc((size_t)192 * 64 * 2);
    _Float16* lin1_h  = (_Float16*)alloc((size_t)64 * 320 * 2);
    float*    qstar   = (float*)   alloc((size_t)G * 128 * 4);
    float*    hs      = (float*)   alloc((size_t)G * 64 * 4);
    float*    cs      = (float*)   alloc((size_t)G * 64 * 4);
    _Float16* feat_h  = (_Float16*)alloc((size_t)T * 320 * 2);
    float*    o1      = (float*)   alloc((size_t)T * 64 * 4);

    auto blks = [](long n, int bs) { return (unsigned)((n + bs - 1) / bs); };
    auto gemm_blocks = [](int M, int Nc) {
        long waves = (long)(M / 16) * (Nc / 64);   // 4 N-tiles per wave
        return (unsigned)((waves * 32 + 255) / 256);
    };

    // weight converts (every call; cheap, no caching)
    k_cvt<<<blks(4096 * 64, 256), 256, 0, stream>>>(e2_w, e2w_h, 4096, 64, 0);
    k_cvt<<<blks(64 * 64, 256), 256, 0, stream>>>(root_w, rootT_h, 64, 64, 1); // out@root_w (not transposed)
    k_cvt<<<blks(192 * 64, 256), 256, 0, stream>>>(gru_wih, wih_h, 192, 64, 0);
    k_cvt<<<blks(192 * 64, 256), 256, 0, stream>>>(gru_whh, whh_h, 192, 64, 0);
    k_cvt<<<blks(64 * 320, 256), 256, 0, stream>>>(lin1_w, lin1_h, 64, 320, 0);

    // front end
    k_lin0<<<blks((long)N * 64, 256), 256, 0, stream>>>(x, lin0_w, lin0_b, node, node_h, N);
    k_e1<<<blks((long)E * 64, 256), 256, 0, stream>>>(edge_attr, e1_w, e1_b, t_h, E);
    k_zero<<<blks(N, 256), 256, 0, stream>>>(deg, N);
    k_count<<<blks(E, 256), 256, 0, stream>>>(dstE, deg, E);

    // ew = t @ e2_w^T + e2_b  -> f16 [E, 4096]   (the 17-GFLOP WMMA GEMM)
    gemm_wmma4<true><<<gemm_blocks(E, 4096), 256, 0, stream>>>(
        t_h, e2w_h, e2_b, ew_h, E, 4096, 64, 0);

    // 6 message-passing + GRU iterations
    for (int it = 0; it < 6; ++it) {
        k_zero<<<blks((long)N * 64, 256), 256, 0, stream>>>(agg, (long)N * 64);
        k_msg<<<blks((long)E * 32, 256), 256, 0, stream>>>(node, ew_h, srcE, dstE, agg, E);
        gemm_wmma4<false><<<gemm_blocks(N, 64), 256, 0, stream>>>(
            node_h, rootT_h, nullptr, rootout, N, 64, 64, 0);
        k_m<<<blks((long)N * 64, 256), 256, 0, stream>>>(agg, deg, rootout, conv_b, m_h, N);
        gemm_wmma4<false><<<gemm_blocks(N, 192), 256, 0, stream>>>(
            m_h, wih_h, gru_bih, gi, N, 192, 64, 0);
        gemm_wmma4<false><<<gemm_blocks(N, 192), 256, 0, stream>>>(
            node_h, whh_h, gru_bhh, gh, N, 192, 64, 0);
        k_gru<<<blks((long)N * 64, 256), 256, 0, stream>>>(gi, gh, node, node_h, N);
    }

    // Set2Set pooling, 6 steps
    k_zero<<<blks((long)G * 128, 256), 256, 0, stream>>>(qstar, (long)G * 128);
    k_zero<<<blks((long)G * 64, 256), 256, 0, stream>>>(hs, (long)G * 64);
    k_zero<<<blks((long)G * 64, 256), 256, 0, stream>>>(cs, (long)G * 64);
    for (int it = 0; it < 6; ++it) {
        k_lstm<<<G, 64, 0, stream>>>(qstar, 128, s2s_wih, s2s_whh, s2s_bih, s2s_bhh, hs, cs, G);
        k_attend<<<G, 64, 0, stream>>>(node, hs, qstar, npg);
    }

    // memory LSTM (one step from zero) -> hx/cx written straight into d_out
    float* out_f = (float*)d_out;
    float* hx = out_f + (size_t)6 * T;
    float* cx = hx + (size_t)G * 64;
    k_zero<<<blks((long)2 * G * 64, 256), 256, 0, stream>>>(hx, (long)2 * G * 64);
    k_lstm<<<G, 64, 0, stream>>>(qstar, 128, mem_wih, mem_whh, mem_bih, mem_bhh, hx, cx, G);

    // head: gather -> lin1 (WMMA, K=320) -> lin2
    k_feat<<<blks((long)T * 320, 256), 256, 0, stream>>>(node, hx, nonring, nrbidx, feat_h, T);
    gemm_wmma4<false><<<gemm_blocks(T, 64), 256, 0, stream>>>(
        feat_h, lin1_h, lin1_b, o1, T, 64, 320, 1);
    k_lin2<<<blks((long)T * 6, 256), 256, 0, stream>>>(o1, lin2_w, lin2_b, out_f, T);
}
